// DCRNN_RecurrentGCN_45801531244822
// MI455X (gfx1250) — matrix operand order repaired and checked by reading, not compiled, not run on hardware
//
#include <hip/hip_runtime.h>
#include <hip/hip_bf16.h>
#include <math.h>

typedef __attribute__((ext_vector_type(16))) _Float16 v16h;
typedef __attribute__((ext_vector_type(8)))  float    v8f;

#define F_IN   256
#define H_DIM  32
#define IN_DIM 288
#define C_OUT  10
#define KSTEPS 9          // 288 / 32
#define ROWS_PER_BLOCK 128  // 8 waves * 16 rows

// ---------------------------------------------------------------------------
// Prep: combine W[0,0]+W[1,0] for the 3 gates and lay out as f16 WMMA
// B-fragments: wfrag[g][s][t][lane][j], lane holds column n = t*16 + (lane&15),
// K = s*32 + (lane>=16 ? 8 : 0) + j for j < 8, s*32 + 16 + (lane>=16 ? 8 : 0) + (j-8) for j >= 8.
// ---------------------------------------------------------------------------
__global__ void dcrnn_prep_weights(const float* __restrict__ Wz,
                                   const float* __restrict__ Wr,
                                   const float* __restrict__ Wh,
                                   _Float16* __restrict__ wfrag) {
    int idx = blockIdx.x * blockDim.x + threadIdx.x;
    const int total = 3 * KSTEPS * 2 * 32 * 16;   // 27648
    if (idx >= total) return;
    int j = idx & 15;
    int l = (idx >> 4) & 31;
    int t = (idx >> 9) & 1;
    int s = (idx >> 10) % KSTEPS;
    int g = idx / (KSTEPS * 2 * 32 * 16);
    const float* W = (g == 0) ? Wz : (g == 1) ? Wr : Wh;
    int K = s * 32 + ((l >= 16) ? 8 : 0) + ((j < 8) ? j : 16 + (j - 8));
    int n = t * 16 + (l & 15);
    // W shape [2,1,288,32]: W[0,0] at 0, W[1,0] at 288*32
    float v = W[(size_t)K * H_DIM + n] + W[(size_t)IN_DIM * H_DIM + (size_t)K * H_DIM + n];
    wfrag[idx] = (_Float16)v;
}

__device__ __forceinline__ float sigmoid_f(float v) {
    return 1.0f / (1.0f + __expf(-v));
}

// Build A fragment (16-bit A 16x32 ISA layout) from 16 f32 values:
// halves 0..7  = contiguous run starting at kOff
// halves 8..15 = contiguous run starting at kOff+16
__device__ __forceinline__ v16h pack_a(const float4 a0, const float4 a1,
                                       const float4 a2, const float4 a3) {
    v16h a;
    a[0] = (_Float16)a0.x; a[1] = (_Float16)a0.y; a[2] = (_Float16)a0.z; a[3] = (_Float16)a0.w;
    a[4] = (_Float16)a1.x; a[5] = (_Float16)a1.y; a[6] = (_Float16)a1.z; a[7] = (_Float16)a1.w;
    a[8]  = (_Float16)a2.x; a[9]  = (_Float16)a2.y; a[10] = (_Float16)a2.z; a[11] = (_Float16)a2.w;
    a[12] = (_Float16)a3.x; a[13] = (_Float16)a3.y; a[14] = (_Float16)a3.z; a[15] = (_Float16)a3.w;
    return a;
}

// ---------------------------------------------------------------------------
// Fused DCRNN cell + tail. One wave32 owns a 16-row tile.
// ---------------------------------------------------------------------------
__global__ __launch_bounds__(256)
void dcrnn_fused_kernel(const float* __restrict__ x,
                        const float* __restrict__ Hst,
                        const _Float16* __restrict__ wfrag,
                        const float* __restrict__ bz,
                        const float* __restrict__ br,
                        const float* __restrict__ bh,
                        const float* __restrict__ Wlin,
                        const float* __restrict__ blin,
                        float* __restrict__ out,
                        int N) {
    __shared__ float s_wlin[H_DIM * C_OUT];
    __shared__ float s_blin[C_OUT];
    __shared__ float s_stage[8][16][33];   // per-wave tile staging (pitch 33 vs bank conflicts)

    const int tid  = threadIdx.x;
    const int wave = tid >> 5;
    const int lane = tid & 31;

    if (tid < H_DIM * C_OUT) s_wlin[tid] = Wlin[tid];
    if (tid < C_OUT)         s_blin[tid] = blin[tid];
    __syncthreads();

    const int rowBase = (blockIdx.x * 8 + wave) * 16;
    const int m  = lane & 15;          // A-fragment row within tile
    const int hi = lane >> 4;          // lane half select
    const int b  = hi * 8;             // K sub-block offset per A layout

    long rowA = (long)rowBase + m;
    if (rowA >= N) rowA = N - 1;       // clamp (stores guarded later)
    const float* xrow = x + rowA * (long)F_IN;
    const float* hrow = Hst + rowA * (long)H_DIM;

    v8f accz0 = {}, accz1 = {}, accr0 = {}, accr1 = {}, acch0 = {}, acch1 = {};

    auto ldb = [&](int g, int s, int t) -> v16h {
        return *(const v16h*)(wfrag + (size_t)((((g * KSTEPS + s) * 2 + t) * 32 + lane) << 4));
    };

    // ---- 8 K-steps over x (shared A fragment feeds all 3 gates) ----
    #pragma unroll
    for (int s = 0; s < 8; ++s) {
        const float4* p0 = (const float4*)(xrow + s * 32 + b);
        const float4* p1 = (const float4*)(xrow + s * 32 + b + 16);
        v16h a = pack_a(p0[0], p0[1], p1[0], p1[1]);
        accz0 = __builtin_amdgcn_wmma_f32_16x16x32_f16(false, a, false, ldb(0, s, 0), (short)0, accz0, false, false);
        accz1 = __builtin_amdgcn_wmma_f32_16x16x32_f16(false, a, false, ldb(0, s, 1), (short)0, accz1, false, false);
        accr0 = __builtin_amdgcn_wmma_f32_16x16x32_f16(false, a, false, ldb(1, s, 0), (short)0, accr0, false, false);
        accr1 = __builtin_amdgcn_wmma_f32_16x16x32_f16(false, a, false, ldb(1, s, 1), (short)0, accr1, false, false);
        acch0 = __builtin_amdgcn_wmma_f32_16x16x32_f16(false, a, false, ldb(2, s, 0), (short)0, acch0, false, false);
        acch1 = __builtin_amdgcn_wmma_f32_16x16x32_f16(false, a, false, ldb(2, s, 1), (short)0, acch1, false, false);
    }

    // ---- K-step 8 (H part) for Z and R ----
    float4 h0 = *(const float4*)(hrow + b);
    float4 h1 = *(const float4*)(hrow + b + 4);
    float4 h2 = *(const float4*)(hrow + b + 16);
    float4 h3 = *(const float4*)(hrow + b + 20);
    {
        v16h ah = pack_a(h0, h1, h2, h3);
        accz0 = __builtin_amdgcn_wmma_f32_16x16x32_f16(false, ah, false, ldb(0, 8, 0), (short)0, accz0, false, false);
        accz1 = __builtin_amdgcn_wmma_f32_16x16x32_f16(false, ah, false, ldb(0, 8, 1), (short)0, accz1, false, false);
        accr0 = __builtin_amdgcn_wmma_f32_16x16x32_f16(false, ah, false, ldb(1, 8, 0), (short)0, accr0, false, false);
        accr1 = __builtin_amdgcn_wmma_f32_16x16x32_f16(false, ah, false, ldb(1, 8, 1), (short)0, accr1, false, false);
    }

    // ---- gates: bias + sigmoid; stage R tile to LDS (D layout -> [row][col]) ----
    float* st = &s_stage[wave][0][0];
    const float bzlo = bz[m],      bzhi = bz[m + 16];
    const float brlo = br[m],      brhi = br[m + 16];
    const float bhlo = bh[m],      bhhi = bh[m + 16];

    #pragma unroll
    for (int i = 0; i < 8; ++i) {
        accz0[i] = sigmoid_f(accz0[i] + bzlo);
        accz1[i] = sigmoid_f(accz1[i] + bzhi);
        float r0 = sigmoid_f(accr0[i] + brlo);
        float r1 = sigmoid_f(accr1[i] + brhi);
        int rowD = i + 8 * hi;                 // D layout: lanes 0-15 -> M=i, lanes 16-31 -> M=8+i
        st[rowD * 33 + m]      = r0;           // col = lane&15
        st[rowD * 33 + 16 + m] = r1;           // col = 16 + (lane&15)
    }
    // intra-wave DS ops are in-order; all lanes' R writes precede the reads below

    // ---- K-step 8 (H*R) for H_tilde: rebuild A fragment from H (regs) * R (LDS) ----
    {
        float hv[16] = { h0.x, h0.y, h0.z, h0.w, h1.x, h1.y, h1.z, h1.w,
                         h2.x, h2.y, h2.z, h2.w, h3.x, h3.y, h3.z, h3.w };
        v16h ah;
        #pragma unroll
        for (int e = 0; e < 8; ++e) {
            ah[e]     = (_Float16)(hv[e]     * st[m * 33 + b + e]);
            ah[e + 8] = (_Float16)(hv[e + 8] * st[m * 33 + b + 16 + e]);
        }
        acch0 = __builtin_amdgcn_wmma_f32_16x16x32_f16(false, ah, false, ldb(2, 8, 0), (short)0, acch0, false, false);
        acch1 = __builtin_amdgcn_wmma_f32_16x16x32_f16(false, ah, false, ldb(2, 8, 1), (short)0, acch1, false, false);
    }

    // ---- GRU combine: H_new = Z*H + (1-Z)*tanh(h~), relu, stage to LDS ----
    #pragma unroll
    for (int i = 0; i < 8; ++i) {
        int rowD = i + 8 * hi;
        long rr = (long)rowBase + rowD;
        if (rr >= N) rr = N - 1;
        float Hlo = Hst[rr * H_DIM + m];
        float Hhi = Hst[rr * H_DIM + 16 + m];
        float htl = tanhf(acch0[i] + bhlo);
        float hth = tanhf(acch1[i] + bhhi);
        float zl = accz0[i], zh = accz1[i];
        float nl = fmaxf(zl * Hlo + (1.0f - zl) * htl, 0.0f);
        float nh = fmaxf(zh * Hhi + (1.0f - zh) * hth, 0.0f);
        st[rowD * 33 + m]      = nl;
        st[rowD * 33 + 16 + m] = nh;
    }
    __syncthreads();

    // ---- tail: per-row linear [32x10] + softmax (lanes 0..15, one row each) ----
    if (lane < 16) {
        long rr = (long)rowBase + lane;
        if (rr < N) {
            float logit[C_OUT];
            #pragma unroll
            for (int c = 0; c < C_OUT; ++c) logit[c] = s_blin[c];
            #pragma unroll
            for (int k = 0; k < H_DIM; ++k) {
                float hvk = st[lane * 33 + k];
                #pragma unroll
                for (int c = 0; c < C_OUT; ++c) logit[c] += hvk * s_wlin[k * C_OUT + c];
            }
            float mx = logit[0];
            #pragma unroll
            for (int c = 1; c < C_OUT; ++c) mx = fmaxf(mx, logit[c]);
            float sum = 0.0f;
            #pragma unroll
            for (int c = 0; c < C_OUT; ++c) { float e = __expf(logit[c] - mx); logit[c] = e; sum += e; }
            float inv = 1.0f / sum;
            #pragma unroll
            for (int c = 0; c < C_OUT; ++c) out[rr * C_OUT + c] = logit[c] * inv;
        }
    }
}

extern "C" void kernel_launch(void* const* d_in, const int* in_sizes, int n_in,
                              void* d_out, int out_size, void* d_ws, size_t ws_size,
                              hipStream_t stream) {
    // setup_inputs order:
    // 0:x [N,256] f32, 1:edge_index (unused), 2:edge_weight (unused), 3:H [N,32] f32,
    // 4:W_z [2,1,288,32], 5:b_z, 6:W_r, 7:b_r, 8:W_h, 9:b_h, 10:W_lin [32,10], 11:b_lin
    const float* x    = (const float*)d_in[0];
    const float* Hst  = (const float*)d_in[3];
    const float* Wz   = (const float*)d_in[4];
    const float* bz   = (const float*)d_in[5];
    const float* Wr   = (const float*)d_in[6];
    const float* br   = (const float*)d_in[7];
    const float* Wh   = (const float*)d_in[8];
    const float* bh   = (const float*)d_in[9];
    const float* Wlin = (const float*)d_in[10];
    const float* blin = (const float*)d_in[11];
    float* out = (float*)d_out;

    const int N = in_sizes[0] / F_IN;   // 200000

    _Float16* wfrag = (_Float16*)d_ws;  // 3*9*2*32*16 halves = 55296 bytes

    const int prep_total = 3 * KSTEPS * 2 * 32 * 16;
    dcrnn_prep_weights<<<(prep_total + 255) / 256, 256, 0, stream>>>(Wz, Wr, Wh, wfrag);

    const int blocks = (N + ROWS_PER_BLOCK - 1) / ROWS_PER_BLOCK;
    dcrnn_fused_kernel<<<blocks, 256, 0, stream>>>(x, Hst, wfrag, bz, br, bh,
                                                   Wlin, blin, out, N);
}
